// DiTBlock_63264868270088
// MI455X (gfx1250) — matrix operand order repaired and checked by reading, not compiled, not run on hardware
//
#include <hip/hip_runtime.h>
#include <hip/hip_bf16.h>

typedef __attribute__((ext_vector_type(16))) __bf16 v16bf;
typedef __attribute__((ext_vector_type(8)))  __bf16 v8bf;
typedef __attribute__((ext_vector_type(8)))  float  v8f;

__device__ inline v8f wmma_bf16(v16bf a, v16bf b, v8f c) {
  // D = A(16x32) * B(32x16) + C, fp32 accumulate
  return __builtin_amdgcn_wmma_f32_16x16x32_bf16(false, a, false, b, (short)0, c, false, false);
}

// ---- cross-lane helpers: butterflies via DPP8 + ds_swizzle ------------------
// DPP8 selector "read lane (i^M)" within each 8-lane group (octal, lane7..lane0)
#define DPP8_XOR1 067452301
#define DPP8_XOR2 054761032
#define DPP8_XOR4 032107654
// ds_swizzle group-of-32: new_lane = ((lane & and) | or) ^ xor
#define SWZ_XOR8  0x201F   // xor=8,  and=0x1f
#define SWZ_XOR16 0x401F   // xor=16, and=0x1f

__device__ inline float xor1_f(float v) {
  return __int_as_float(__builtin_amdgcn_mov_dpp8(__float_as_int(v), DPP8_XOR1));
}
__device__ inline float xor2_f(float v) {
  return __int_as_float(__builtin_amdgcn_mov_dpp8(__float_as_int(v), DPP8_XOR2));
}
__device__ inline float xor4_f(float v) {
  return __int_as_float(__builtin_amdgcn_mov_dpp8(__float_as_int(v), DPP8_XOR4));
}
__device__ inline float xor8_f(float v) {
  return __int_as_float(__builtin_amdgcn_ds_swizzle(__float_as_int(v), SWZ_XOR8));
}
__device__ inline float xor16_f(float v) {
  return __int_as_float(__builtin_amdgcn_ds_swizzle(__float_as_int(v), SWZ_XOR16));
}
__device__ inline float rowmax16(float v) {
  v = fmaxf(v, xor1_f(v));
  v = fmaxf(v, xor2_f(v));
  v = fmaxf(v, xor4_f(v));
  v = fmaxf(v, xor8_f(v));
  return v;
}
__device__ inline float rowsum16(float v) {
  v += xor1_f(v);
  v += xor2_f(v);
  v += xor4_f(v);
  v += xor8_f(v);
  return v;
}
__device__ inline float wavesum32(float v) {
  v += xor1_f(v);
  v += xor2_f(v);
  v += xor4_f(v);
  v += xor8_f(v);
  v += xor16_f(v);
  return v;
}

// ---- fragment loaders (contiguous k order, same for A and B) --------------------------

// A fragment: 16x32 (MxK) from row-major source, row stride lda (elements).
// Lane l: M = l&15, kh = l>>4; elems j=0..15 -> K = kh*16+j
__device__ inline v16bf load_a16x32(const __bf16* A, int lda, int row0, int k0, int lane) {
  const int m  = lane & 15;
  const int kh = lane >> 4;
  const __bf16* p = A + (size_t)(row0 + m) * lda + k0 + kh * 16;
  union { v8bf h[2]; v16bf v; } u;
  u.h[0] = *(const v8bf*)(p);
  u.h[1] = *(const v8bf*)(p + 8);
  return u.v;
}

// B fragment: 32x16 (KxN) where source T is laid out T[n][k] row-major (stride ldk).
// Lane l: N = l&15, kh = l>>4; elems j=0..15 -> K = kh*16+j  (32 contiguous bytes)
__device__ inline v16bf load_b32x16_T(const __bf16* T, int ldk, int n0, int k0, int lane) {
  const int n  = lane & 15;
  const int kh = lane >> 4;
  const __bf16* p = T + (size_t)(n0 + n) * ldk + k0 + kh * 16;
  union { v8bf h[2]; v16bf v; } u;
  u.h[0] = *(const v8bf*)(p);
  u.h[1] = *(const v8bf*)(p + 8);
  return u.v;
}

// scale every element of a bf16 fragment (exact for powers of two)
__device__ inline v16bf scale_frag(v16bf a, float s) {
  union { v16bf v; __bf16 e[16]; } u;
  u.v = a;
  #pragma unroll
  for (int j = 0; j < 16; ++j) u.e[j] = (__bf16)((float)u.e[j] * s);
  return u.v;
}

// ---- AdaLN head: h = silu(cond) @ W + b  (tiny GEMV) ------------------------
__global__ void k_adaln(const float* __restrict__ cond, const float* __restrict__ W,
                        const float* __restrict__ b, float* __restrict__ hout,
                        int D, int N3) {
  int j = blockIdx.x * blockDim.x + threadIdx.x;
  if (j >= N3) return;
  float acc = b[j];
  for (int i = 0; i < D; ++i) {
    float c = cond[i];
    float sv = c / (1.0f + __expf(-c));   // silu
    acc += sv * W[(size_t)i * N3 + j];
  }
  hout[j] = acc;
}

// ---- tiled weight convert fp32 -> bf16, transposed: WT[n][k] = W[k][n] ------
__global__ __launch_bounds__(256) void k_cvt_t(const float* __restrict__ W,
                                               __bf16* __restrict__ WT, int K, int N) {
  __shared__ __bf16 tile[32][33];
  const int k0 = blockIdx.x * 32;
  const int n0 = blockIdx.y * 32;
  const int tx = threadIdx.x & 31;
  const int ty = threadIdx.x >> 5;
  #pragma unroll
  for (int i = ty; i < 32; i += 8)
    tile[i][tx] = (__bf16)W[(size_t)(k0 + i) * N + n0 + tx];   // coalesced read
  __syncthreads();
  #pragma unroll
  for (int i = ty; i < 32; i += 8)
    WT[(size_t)(n0 + i) * K + k0 + tx] = tile[tx][i];          // coalesced write
}

// ---- tiled V transpose: Vt[d][l] = qkv[l][1536 + d]  (bf16) -----------------
__global__ __launch_bounds__(256) void k_vt(const __bf16* __restrict__ qkv,
                                            __bf16* __restrict__ Vt, int L) {
  __shared__ __bf16 tile[32][33];
  const int l0 = blockIdx.x * 32;
  const int d0 = blockIdx.y * 32;
  const int tx = threadIdx.x & 31;
  const int ty = threadIdx.x >> 5;
  #pragma unroll
  for (int i = ty; i < 32; i += 8)
    tile[i][tx] = qkv[(size_t)(l0 + i) * 2304 + 1536 + d0 + tx];
  __syncthreads();
  #pragma unroll
  for (int i = ty; i < 32; i += 8)
    Vt[(size_t)(d0 + i) * L + l0 + tx] = tile[tx][i];
}

// ---- fused LayerNorm + modulate -> bf16 -------------------------------------
__global__ __launch_bounds__(256) void k_ln_mod(const float* __restrict__ x,
                                                const float* __restrict__ h,  // [3*768]: shift|scale|gate
                                                __bf16* __restrict__ xn) {
  const int row = blockIdx.x;
  const float* xr = x + (size_t)row * 768;
  __shared__ float redS[8], redQ[8];
  float s = 0.f, ss = 0.f;
  for (int c = threadIdx.x; c < 768; c += 256) { float v = xr[c]; s += v; ss += v * v; }
  s  = wavesum32(s);
  ss = wavesum32(ss);
  if ((threadIdx.x & 31) == 0) { redS[threadIdx.x >> 5] = s; redQ[threadIdx.x >> 5] = ss; }
  __syncthreads();
  s = 0.f; ss = 0.f;
  #pragma unroll
  for (int i = 0; i < 8; ++i) { s += redS[i]; ss += redQ[i]; }
  const float mu = s * (1.0f / 768.0f);
  const float var = ss * (1.0f / 768.0f) - mu * mu;
  const float rstd = rsqrtf(var + 1e-5f);
  for (int c = threadIdx.x; c < 768; c += 256) {
    float v = (xr[c] - mu) * rstd;
    xn[(size_t)row * 768 + c] = (__bf16)(v * (1.0f + h[768 + c]) + h[c]);
  }
}

// ---- WMMA GEMM: C = A[MxK] * BT^T + bias, fused epilogues -------------------
// Block: 128 threads = 4 waves in 2x2; wave tile 32x64 (2x4 fragments);
// block tile 64 rows x 128 cols.
enum { EPI_BF16 = 0, EPI_GELU_BF16 = 1, EPI_GATE_RES_F32 = 2 };

template <int EPI>
__global__ __launch_bounds__(128) void k_gemm_bf16(
    const __bf16* __restrict__ A, const __bf16* __restrict__ BT,
    const float* __restrict__ bias, int M, int N, int Kd,
    float* __restrict__ Cf, __bf16* __restrict__ Cb,
    const float* __restrict__ res, const float* __restrict__ gate) {
  const int lane = threadIdx.x & 31;
  const int wave = threadIdx.x >> 5;
  const int row0 = blockIdx.y * 64  + (wave >> 1) * 32;
  const int col0 = blockIdx.x * 128 + (wave & 1) * 64;

  v8f acc[2][4] = {};
  for (int k0 = 0; k0 < Kd; k0 += 32) {
    // prefetch next k-slice into the near (WGP) cache while this slice computes
    if (k0 + 32 < Kd) {
      __builtin_prefetch(A  + (size_t)(row0 + lane) * Kd + k0 + 32, 0, 3);
      __builtin_prefetch(BT + (size_t)(col0 + lane) * Kd + k0 + 32, 0, 3);
      __builtin_prefetch(BT + (size_t)(col0 + 32 + lane) * Kd + k0 + 32, 0, 3);
    }
    v16bf a0 = load_a16x32(A, Kd, row0,      k0, lane);
    v16bf a1 = load_a16x32(A, Kd, row0 + 16, k0, lane);
    v16bf b0 = load_b32x16_T(BT, Kd, col0,      k0, lane);
    v16bf b1 = load_b32x16_T(BT, Kd, col0 + 16, k0, lane);
    v16bf b2 = load_b32x16_T(BT, Kd, col0 + 32, k0, lane);
    v16bf b3 = load_b32x16_T(BT, Kd, col0 + 48, k0, lane);
    acc[0][0] = wmma_bf16(a0, b0, acc[0][0]);
    acc[0][1] = wmma_bf16(a0, b1, acc[0][1]);
    acc[0][2] = wmma_bf16(a0, b2, acc[0][2]);
    acc[0][3] = wmma_bf16(a0, b3, acc[0][3]);
    acc[1][0] = wmma_bf16(a1, b0, acc[1][0]);
    acc[1][1] = wmma_bf16(a1, b1, acc[1][1]);
    acc[1][2] = wmma_bf16(a1, b2, acc[1][2]);
    acc[1][3] = wmma_bf16(a1, b3, acc[1][3]);
  }

  const int nl = lane & 15;
  const int g  = lane >> 4;
  #pragma unroll
  for (int fi = 0; fi < 2; ++fi) {
    #pragma unroll
    for (int fj = 0; fj < 4; ++fj) {
      const int col = col0 + fj * 16 + nl;
      const float bv = bias[col];
      float gv = 0.f;
      if (EPI == EPI_GATE_RES_F32) gv = gate[col];
      #pragma unroll
      for (int r = 0; r < 8; ++r) {
        const int row = row0 + fi * 16 + 8 * g + r;
        const size_t idx = (size_t)row * N + col;
        float v = acc[fi][fj][r] + bv;
        if (EPI == EPI_BF16) {
          Cb[idx] = (__bf16)v;
        } else if (EPI == EPI_GELU_BF16) {
          Cb[idx] = (__bf16)(0.5f * v * (1.0f + erff(v * 0.70710678118654752f)));
        } else {
          Cf[idx] = res[idx] + gv * v;
        }
      }
    }
  }
}

// ---- Flash attention: one wave per 16 query rows, 64-key tiles --------------
__global__ __launch_bounds__(128) void k_flash_attn(const __bf16* __restrict__ qkv,
                                                    const __bf16* __restrict__ Vt,  // [768][L]
                                                    __bf16* __restrict__ O, int L) {
  const int lane = threadIdx.x & 31;
  const int wave = threadIdx.x >> 5;
  const int h    = blockIdx.y;
  const int qr   = blockIdx.x * 64 + wave * 16;
  const int LDQ  = 2304;

  const __bf16* Q   = qkv + h * 64;
  const __bf16* K   = qkv + 768 + h * 64;
  const __bf16* VtH = Vt + (size_t)h * 64 * L;   // rows = head-dim, cols = keys

  __shared__ __align__(16) __bf16 Pbuf[4 * 16 * 64];
  __bf16* Pw = Pbuf + wave * (16 * 64);

  // Q pre-scaled by 1/sqrt(hd) = 0.125 (exact in bf16)
  const v16bf aq0 = scale_frag(load_a16x32(Q, LDQ, qr, 0,  lane), 0.125f);
  const v16bf aq1 = scale_frag(load_a16x32(Q, LDQ, qr, 32, lane), 0.125f);

  v8f o[4] = {};
  float mrow[8], lrow[8];
  #pragma unroll
  for (int r = 0; r < 8; ++r) { mrow[r] = -1e30f; lrow[r] = 0.f; }

  const int nl = lane & 15;
  const int g  = lane >> 4;

  for (int kt = 0; kt < L; kt += 64) {
    // prefetch next K / V^T tile into the near cache (speculative at the end)
    __builtin_prefetch(K   + (size_t)(kt + 64 + lane) * LDQ, 0, 3);
    __builtin_prefetch(K   + (size_t)(kt + 96 + lane) * LDQ, 0, 3);
    __builtin_prefetch(VtH + (size_t)lane * L + kt + 64, 0, 3);
    __builtin_prefetch(VtH + (size_t)(32 + lane) * L + kt + 64, 0, 3);

    // S = Q * K^T for 64 keys -> four 16x16 score fragments (pre-scaled)
    v8f s[4] = {};
    #pragma unroll
    for (int j = 0; j < 4; ++j) {
      s[j] = wmma_bf16(aq0, load_b32x16_T(K, LDQ, kt + j * 16, 0,  lane), s[j]);
      s[j] = wmma_bf16(aq1, load_b32x16_T(K, LDQ, kt + j * 16, 32, lane), s[j]);
    }

    float ps[4][8];
    #pragma unroll
    for (int r = 0; r < 8; ++r) {
      float mx = fmaxf(fmaxf(s[0][r], s[1][r]), fmaxf(s[2][r], s[3][r]));
      mx = rowmax16(mx);
      const float mnew  = fmaxf(mrow[r], mx);
      const float alpha = __expf(mrow[r] - mnew);
      mrow[r] = mnew;
      #pragma unroll
      for (int j = 0; j < 4; ++j) ps[j][r] = __expf(s[j][r] - mnew);
      const float rs = rowsum16((ps[0][r] + ps[1][r]) + (ps[2][r] + ps[3][r]));
      lrow[r] = lrow[r] * alpha + rs;
      #pragma unroll
      for (int t = 0; t < 4; ++t) o[t][r] *= alpha;
    }

    // stage P (C layout -> row-major 16x64 bf16 in LDS), then reload as A frags
    #pragma unroll
    for (int r = 0; r < 8; ++r) {
      #pragma unroll
      for (int j = 0; j < 4; ++j)
        Pw[(8 * g + r) * 64 + j * 16 + nl] = (__bf16)ps[j][r];
    }
    asm volatile("s_wait_dscnt 0" ::: "memory");  // intra-wave RAW through LDS
    const v16bf ap0 = load_a16x32(Pw, 64, 0, 0,  lane);
    const v16bf ap1 = load_a16x32(Pw, 64, 0, 32, lane);

    // O += P * V : B fragments from V^T are contiguous 32B per lane
    #pragma unroll
    for (int t = 0; t < 4; ++t) {
      o[t] = wmma_bf16(ap0, load_b32x16_T(VtH, L, t * 16, kt,      lane), o[t]);
      o[t] = wmma_bf16(ap1, load_b32x16_T(VtH, L, t * 16, kt + 32, lane), o[t]);
    }
  }

  float inv[8];
  #pragma unroll
  for (int r = 0; r < 8; ++r) inv[r] = 1.0f / lrow[r];
  #pragma unroll
  for (int r = 0; r < 8; ++r) {
    const size_t base = (size_t)(qr + 8 * g + r) * 768 + h * 64 + nl;
    #pragma unroll
    for (int t = 0; t < 4; ++t)
      O[base + t * 16] = (__bf16)(o[t][r] * inv[r]);
  }
}

// ---- launch -----------------------------------------------------------------
extern "C" void kernel_launch(void* const* d_in, const int* in_sizes, int n_in,
                              void* d_out, int out_size, void* d_ws, size_t ws_size,
                              hipStream_t stream) {
  const float* x        = (const float*)d_in[0];
  const float* cond     = (const float*)d_in[1];
  const float* w_adaln1 = (const float*)d_in[2];
  const float* b_adaln1 = (const float*)d_in[3];
  const float* w_qkv    = (const float*)d_in[4];
  const float* b_qkv    = (const float*)d_in[5];
  const float* w_attn   = (const float*)d_in[6];
  const float* b_attn   = (const float*)d_in[7];
  const float* w_adaln2 = (const float*)d_in[8];
  const float* b_adaln2 = (const float*)d_in[9];
  const float* w_ffn1   = (const float*)d_in[10];
  const float* b_ffn1   = (const float*)d_in[11];
  const float* w_ffn2   = (const float*)d_in[12];
  const float* b_ffn2   = (const float*)d_in[13];
  float* out = (float*)d_out;

  const int L = 4096, D = 768, H = 12, D3 = 2304, Dm = 3072;

  char* ws = (char*)d_ws;
  size_t off = 0;
  auto alloc = [&](size_t bytes) -> void* {
    void* p = ws + off;
    off += (bytes + 255) & ~(size_t)255;
    return p;
  };
  float*  h1     = (float*) alloc((size_t)D3 * 4);
  float*  h2     = (float*) alloc((size_t)D3 * 4);
  __bf16* wqkvT  = (__bf16*)alloc((size_t)D  * D3 * 2);
  __bf16* wattnT = (__bf16*)alloc((size_t)D  * D  * 2);
  __bf16* wffn1T = (__bf16*)alloc((size_t)D  * Dm * 2);
  __bf16* wffn2T = (__bf16*)alloc((size_t)Dm * D  * 2);
  __bf16* xn     = (__bf16*)alloc((size_t)L  * D  * 2);
  __bf16* qkv    = (__bf16*)alloc((size_t)L  * D3 * 2);
  __bf16* Vt     = (__bf16*)alloc((size_t)L  * D  * 2);
  __bf16* attnO  = (__bf16*)alloc((size_t)L  * D  * 2);
  float*  x1     = (float*) alloc((size_t)L  * D  * 4);
  __bf16* hmid   = (__bf16*)alloc((size_t)L  * Dm * 2);

  // AdaLN heads (both depend only on cond)
  k_adaln<<<dim3((D3 + 255) / 256), 256, 0, stream>>>(cond, w_adaln1, b_adaln1, h1, D, D3);
  k_adaln<<<dim3((D3 + 255) / 256), 256, 0, stream>>>(cond, w_adaln2, b_adaln2, h2, D, D3);

  // weight conversions (fp32 -> bf16, tiled transpose)
  k_cvt_t<<<dim3(D / 32,  D3 / 32), 256, 0, stream>>>(w_qkv,  wqkvT,  D,  D3);
  k_cvt_t<<<dim3(D / 32,  D  / 32), 256, 0, stream>>>(w_attn, wattnT, D,  D);
  k_cvt_t<<<dim3(D / 32,  Dm / 32), 256, 0, stream>>>(w_ffn1, wffn1T, D,  Dm);
  k_cvt_t<<<dim3(Dm / 32, D  / 32), 256, 0, stream>>>(w_ffn2, wffn2T, Dm, D);

  // attention branch
  k_ln_mod<<<dim3(L), 256, 0, stream>>>(x, h1, xn);
  k_gemm_bf16<EPI_BF16><<<dim3(D3 / 128, L / 64), 128, 0, stream>>>(
      xn, wqkvT, b_qkv, L, D3, D, nullptr, qkv, nullptr, nullptr);
  k_vt<<<dim3(L / 32, D / 32), 256, 0, stream>>>(qkv, Vt, L);
  k_flash_attn<<<dim3(L / 64, H), 128, 0, stream>>>(qkv, Vt, attnO, L);
  k_gemm_bf16<EPI_GATE_RES_F32><<<dim3(D / 128, L / 64), 128, 0, stream>>>(
      attnO, wattnT, b_attn, L, D, D, x1, nullptr, x, h1 + 2 * D);

  // FFN branch
  k_ln_mod<<<dim3(L), 256, 0, stream>>>(x1, h2, xn);
  k_gemm_bf16<EPI_GELU_BF16><<<dim3(Dm / 128, L / 64), 128, 0, stream>>>(
      xn, wffn1T, b_ffn1, L, Dm, D, nullptr, hmid, nullptr, nullptr);
  k_gemm_bf16<EPI_GATE_RES_F32><<<dim3(D / 128, L / 64), 128, 0, stream>>>(
      hmid, wffn2T, b_ffn2, L, D, Dm, out, nullptr, x1, h2 + 2 * D);

  (void)in_sizes; (void)n_in; (void)out_size; (void)ws_size; (void)H;
}
